// MultiHeadAttentionCNN_67989332295754
// MI455X (gfx1250) — compile-verified, matrix operand order checked by reading, not run on hardware
//
#include <hip/hip_runtime.h>
#include <hip/hip_bf16.h>

// ---------------- types ----------------
typedef __bf16 bf16_t;
typedef __attribute__((ext_vector_type(16))) __bf16 v16bf;
typedef __attribute__((ext_vector_type(8)))  __bf16 v8bf;
typedef __attribute__((ext_vector_type(4)))  __bf16 v4bf;
typedef __attribute__((ext_vector_type(8)))  float  v8f;
typedef __attribute__((ext_vector_type(4)))  float  v4f;
typedef __attribute__((ext_vector_type(4)))  int    v4i;

// address-space qualified pointer types (AS1 = global, AS3 = LDS)
typedef const __attribute__((address_space(1))) bf16_t* gcbf_t;
typedef const __attribute__((address_space(3))) bf16_t* lcbf_t;
typedef __attribute__((address_space(1))) bf16_t*       gbf_t;
typedef __attribute__((address_space(3))) bf16_t*       lbf_t;
typedef __attribute__((address_space(1))) float*        gf_t;
typedef const __attribute__((address_space(1))) v8bf*   gv8bf_cp;
typedef const __attribute__((address_space(3))) v8bf*   lv8bf_cp;
typedef __attribute__((address_space(3))) v8bf*         lv8bf_p;
typedef const __attribute__((address_space(1))) v4f*    gv4f_cp;
typedef __attribute__((address_space(3))) v4bf*         lv4bf_p;

#define BATCH 8
#define DIM   384
#define NH    8
#define HDIM  48
#define HDP   64      // head dim padded to 2 x K32 WMMA chunks
#define SEQ   1024
#define QSCALE 0.14433756729740643f  // 48^-0.5

// ---------------- async copy to LDS (ASYNCcnt path, guarded) ----------------
#if defined(__HIP_DEVICE_COMPILE__) && __has_builtin(__builtin_amdgcn_global_load_async_to_lds_b128)
#define USE_ASYNC_LDS 1
#else
#define USE_ASYNC_LDS 0
#endif

#if USE_ASYNC_LDS
typedef __attribute__((address_space(1))) v4i* gv4p_t;
typedef __attribute__((address_space(3))) v4i* lv4p_t;
#endif

__device__ __forceinline__ void cp16_to_lds(void* l, const void* g) {
#if USE_ASYNC_LDS
  __builtin_amdgcn_global_load_async_to_lds_b128((gv4p_t)g, (lv4p_t)l, 0, 0);
#else
  *(v4f*)l = *(const v4f*)g;
#endif
}

__device__ __forceinline__ void cp_wait_all() {
#if USE_ASYNC_LDS && __has_builtin(__builtin_amdgcn_s_wait_asynccnt)
  __builtin_amdgcn_s_wait_asynccnt(0);
#endif
}

// ---------------- Tensor Data Mover (TENSORcnt path, guarded) ---------------
#if defined(__HIP_DEVICE_COMPILE__) && __has_builtin(__builtin_amdgcn_tensor_load_to_lds) && __has_builtin(__builtin_amdgcn_s_wait_tensorcnt)
#define USE_TDM 1
#else
#define USE_TDM 0
#endif

#if USE_TDM
typedef __attribute__((ext_vector_type(4))) unsigned int u32x4;
typedef __attribute__((ext_vector_type(8))) int i32x8;
typedef __attribute__((ext_vector_type(4))) int i32x4;

// 2D tile DMA: global (row-major, stride0 elements) -> LDS (tile rows packed).
// D# per CDNA5 ISA ch.8: group0 = {count, lds_addr, global_addr, type=2},
// group1 = {data_size=2B, tensor_dim0/1, tile_dim0/1, tensor_dim0_stride}.
__device__ __forceinline__ void tdm_load_2d(const void* g, unsigned lds_byte_off,
                                            int tensor_d0, int tensor_d1,
                                            int tile_d0, int tile_d1,
                                            int stride0) {
  const unsigned long long ga = (unsigned long long)g;
  u32x4 g0;
  g0[0] = 1u;                                   // count=1 (user descriptor)
  g0[1] = lds_byte_off;                         // lds_addr (bytes)
  g0[2] = (unsigned)ga;                         // global_addr[31:0]
  g0[3] = (unsigned)((ga >> 32) & 0x1FFFFFFull) // global_addr[56:32]
          | 0x80000000u;                        // type = 2 ("image")
  i32x8 g1;
  g1[0] = 1 << 16;          // workgroup_mask=0 | data_size=1 (2 bytes)
  g1[1] = tensor_d0 << 16;  // tensor_dim0[15:0] -> bits 63:48
  g1[2] = tensor_d1 << 16;  // tensor_dim0 hi=0 | tensor_dim1[15:0]
  g1[3] = tile_d0 << 16;    // tensor_dim1 hi=0 | tile_dim0
  g1[4] = tile_d1 & 0xFFFF; // tile_dim1 | tile_dim2=0
  g1[5] = stride0;          // tensor_dim0_stride[31:0]
  g1[6] = 0;                // stride0 hi | tensor_dim1_stride lo (unused, 2D)
  g1[7] = 0;
  const i32x4 z4 = {0, 0, 0, 0};
  const i32x8 z8 = {0, 0, 0, 0, 0, 0, 0, 0};
  __builtin_amdgcn_tensor_load_to_lds(g0, g1, z4, z4, z8, 0);
}
#endif

// ---------------- WMMA helpers ----------------
// A-fragment (16x32 bf16): lane L: row = L%16, hi = L/16; lane elements map to
// k = {8*hi..8*hi+7} and {16+8*hi..23+8*hi} -> two contiguous 16 B loads.
// The same loader reads a B-fragment from the transposed source (row n = B[:,n]).
__device__ __forceinline__ v16bf load_frag_g(gcbf_t base, int ld, int k0) {
  const int lane = (int)(threadIdx.x & 31u);
  const int r = lane & 15, hi = lane >> 4;
  gcbf_t p = base + r * ld + k0 + 8 * hi;
  v8bf lo = *(gv8bf_cp)p;
  v8bf hh = *(gv8bf_cp)(p + 16);
  v16bf out;
#pragma unroll
  for (int i = 0; i < 8; ++i) { out[i] = lo[i]; out[i + 8] = hh[i]; }
  return out;
}

__device__ __forceinline__ v16bf load_frag_l(lcbf_t base, int ld, int k0) {
  const int lane = (int)(threadIdx.x & 31u);
  const int r = lane & 15, hi = lane >> 4;
  lcbf_t p = base + r * ld + k0 + 8 * hi;
  v8bf lo = *(lv8bf_cp)p;
  v8bf hh = *(lv8bf_cp)(p + 16);
  v16bf out;
#pragma unroll
  for (int i = 0; i < 8; ++i) { out[i] = lo[i]; out[i + 8] = hh[i]; }
  return out;
}

__device__ __forceinline__ v8f wmma_bf16(v16bf a, v16bf b, v8f c) {
  return __builtin_amdgcn_wmma_f32_16x16x32_bf16(false, a, false, b, (short)0, c,
                                                 false, false);
}

// ---------------- small utility kernels ----------------
__global__ void fill_zero_kernel(unsigned int* p, size_t n32) {
  size_t i = (size_t)blockIdx.x * blockDim.x + threadIdx.x;
  size_t stride = (size_t)gridDim.x * blockDim.x;
  for (; i < n32; i += stride) ((__attribute__((address_space(1))) unsigned int*)p)[i] = 0u;
}

__global__ void cvt_kernel(const float* __restrict__ s, bf16_t* __restrict__ d, int n) {
  int i = blockIdx.x * blockDim.x + threadIdx.x;
  int stride = gridDim.x * blockDim.x;
  for (; i < n; i += stride)
    ((gbf_t)d)[i] = (bf16_t)((const __attribute__((address_space(1))) float*)s)[i];
}

// ---------------- kernel 1: fused QKV projection + head split ----------------
__global__ __launch_bounds__(256) void qkv_kernel(
    const float* __restrict__ x,
    const bf16_t* __restrict__ wqb, const bf16_t* __restrict__ wkb,
    const bf16_t* __restrict__ wvb,
    const float* __restrict__ bq, const float* __restrict__ bk,
    const float* __restrict__ bv,
    bf16_t* __restrict__ Qp, bf16_t* __restrict__ Kp, bf16_t* __restrict__ VpT) {
  __shared__ __attribute__((aligned(16))) bf16_t xs[64 * DIM];  // 48 KB
  const int b  = blockIdx.y;
  const int t0 = blockIdx.x * 64;
  const int tid = threadIdx.x;

  // stage x tile fp32 -> bf16 (tile is contiguous in memory)
  const float* xsrc = x + ((size_t)b * SEQ + t0) * DIM;
  lbf_t xsl = (lbf_t)xs;
  for (int e = tid * 4; e < 64 * DIM; e += 256 * 4) {
    v4f f = *(gv4f_cp)(xsrc + e);
    v4bf h2;
    h2[0] = (bf16_t)f[0]; h2[1] = (bf16_t)f[1];
    h2[2] = (bf16_t)f[2]; h2[3] = (bf16_t)f[3];
    *(lv4bf_p)(xsl + e) = h2;
  }
  __syncthreads();

  const int wave = tid >> 5, lane = tid & 31;
  const int n = lane & 15, hi = lane >> 4;
  const bf16_t* Wsel[3] = {wqb, wkb, wvb};
  const float*  Bsel[3] = {bq, bk, bv};

  for (int job = wave; job < 4 * 72; job += 8) {
    const int mt  = job & 3;
    const int gn  = (job >> 2) * 16;    // 0..1151
    const int id  = gn / DIM;           // 0=q 1=k 2=v
    const int nin = gn % DIM;
    gcbf_t wb = (gcbf_t)(Wsel[id] + (size_t)nin * DIM);  // B^T rows == W rows
    v8f acc = {0.f, 0.f, 0.f, 0.f, 0.f, 0.f, 0.f, 0.f};
#pragma unroll
    for (int kk = 0; kk < DIM; kk += 32) {
      v16bf a  = load_frag_l((lcbf_t)xs + mt * 16 * DIM, DIM, kk);
      v16bf bm = load_frag_g(wb, DIM, kk);
      acc = wmma_bf16(a, bm, acc);
    }
    const float bias = Bsel[id][nin + n];
    const int head = nin / HDIM;
    const int col  = (nin % HDIM) + n;   // 16-tiles never straddle heads
    const int tok0 = t0 + mt * 16;
    if (id == 2) {
      gbf_t dst = (gbf_t)(VpT + (((size_t)b * NH + head) * HDP + col) * SEQ +
                          tok0 + 8 * hi);
#pragma unroll
      for (int i = 0; i < 8; ++i) dst[i] = (bf16_t)(acc[i] + bias);
    } else {
      gbf_t dst = (gbf_t)((id == 0 ? Qp : Kp) +
                          (((size_t)b * NH + head) * SEQ + tok0 + 8 * hi) * HDP +
                          col);
      const float sc = (id == 0) ? QSCALE : 1.0f;
#pragma unroll
      for (int i = 0; i < 8; ++i)
        dst[(size_t)i * HDP] = (bf16_t)((acc[i] + bias) * sc);
    }
  }
}

// ---------------- kernel 2: flash attention ----------------
__global__ __launch_bounds__(256) void attn_kernel(
    const bf16_t* __restrict__ Qp, const bf16_t* __restrict__ Kp,
    const bf16_t* __restrict__ VpT, bf16_t* __restrict__ AO) {
  __shared__ __attribute__((aligned(16))) bf16_t ks[64 * HDP];      // 8 KB
  __shared__ __attribute__((aligned(16))) bf16_t vs[HDP * 64];      // 8 KB
  __shared__ __attribute__((aligned(16))) bf16_t ps[8][16 * 64];    // 16 KB
  const int b = blockIdx.z, h = blockIdx.y;
  const int q0 = blockIdx.x * 128;
  const int tid = threadIdx.x, wave = tid >> 5, lane = tid & 31;
  const int n = lane & 15, hi = lane >> 4;

  const bf16_t* Qb = Qp + ((size_t)b * NH + h) * SEQ * HDP;
  const bf16_t* Kb = Kp + ((size_t)b * NH + h) * SEQ * HDP;
  const bf16_t* Vb = VpT + ((size_t)b * NH + h) * HDP * SEQ;
  const int qrow = q0 + wave * 16;

#if USE_TDM
  const unsigned ks_off = (unsigned)(unsigned long long)(lbf_t)ks;
#endif

  v16bf qa0 = load_frag_g((gcbf_t)(Qb + (size_t)qrow * HDP), HDP, 0);
  v16bf qa1 = load_frag_g((gcbf_t)(Qb + (size_t)qrow * HDP), HDP, 32);

  v8f o[3];
  float mrow[8], lrow[8];
#pragma unroll
  for (int c2 = 0; c2 < 3; ++c2)
#pragma unroll
    for (int i = 0; i < 8; ++i) o[c2][i] = 0.f;
#pragma unroll
  for (int i = 0; i < 8; ++i) { mrow[i] = -1e30f; lrow[i] = 0.f; }

  for (int kc = 0; kc < SEQ; kc += 64) {
    // --- stage K chunk [64 keys][HDP] (contiguous 8 KB): TDM if available ---
#if USE_TDM
    if (wave == 0)
      tdm_load_2d(Kb + (size_t)kc * HDP, ks_off, HDP, 64, HDP, 64, HDP);
#else
    for (int e = tid * 8; e < 64 * HDP; e += 256 * 8)
      cp16_to_lds(ks + e, Kb + (size_t)kc * HDP + e);
#endif
    // --- stage V^T chunk (48 real cols x 64 keys): async-LDS path ---
    for (int cid = tid; cid < HDIM * 8; cid += 256) {
      const int c = cid >> 3, qd = cid & 7;
      cp16_to_lds(vs + c * 64 + qd * 8, Vb + (size_t)c * SEQ + kc + qd * 8);
    }
    cp_wait_all();
#if USE_TDM
    if (wave == 0) __builtin_amdgcn_s_wait_tensorcnt(0);
#endif
    __syncthreads();

    // scores: [16 q][64 keys], 1/sqrt(48) already folded into Q
    v8f st[4];
#pragma unroll
    for (int nt2 = 0; nt2 < 4; ++nt2) {
      v16bf kb0 = load_frag_l((lcbf_t)ks + nt2 * 16 * HDP, HDP, 0);
      v16bf kb1 = load_frag_l((lcbf_t)ks + nt2 * 16 * HDP, HDP, 32);
      v8f s = {0.f, 0.f, 0.f, 0.f, 0.f, 0.f, 0.f, 0.f};
      s = wmma_bf16(qa0, kb0, s);
      s = wmma_bf16(qa1, kb1, s);
      st[nt2] = s;
    }

    // online softmax; row m = i + 8*hi lives across a 16-lane half
    float cm[8];
#pragma unroll
    for (int i = 0; i < 8; ++i)
      cm[i] = fmaxf(fmaxf(st[0][i], st[1][i]), fmaxf(st[2][i], st[3][i]));
#pragma unroll
    for (int msk = 1; msk < 16; msk <<= 1)
#pragma unroll
      for (int i = 0; i < 8; ++i)
        cm[i] = fmaxf(cm[i], __shfl_xor(cm[i], msk, 32));

    float mn[8], scl[8], rs[8];
#pragma unroll
    for (int i = 0; i < 8; ++i) {
      mn[i]  = fmaxf(mrow[i], cm[i]);
      scl[i] = __expf(mrow[i] - mn[i]);
      rs[i]  = 0.f;
    }
#pragma unroll
    for (int nt2 = 0; nt2 < 4; ++nt2)
#pragma unroll
      for (int i = 0; i < 8; ++i) {
        const float p = __expf(st[nt2][i] - mn[i]);
        st[nt2][i] = p;
        rs[i] += p;
      }
#pragma unroll
    for (int msk = 1; msk < 16; msk <<= 1)
#pragma unroll
      for (int i = 0; i < 8; ++i) rs[i] += __shfl_xor(rs[i], msk, 32);
#pragma unroll
    for (int i = 0; i < 8; ++i) {
      lrow[i] = lrow[i] * scl[i] + rs[i];
      mrow[i] = mn[i];
    }
#pragma unroll
    for (int c2 = 0; c2 < 3; ++c2)
#pragma unroll
      for (int i = 0; i < 8; ++i) o[c2][i] *= scl[i];

    // C-layout -> A-layout for P via wave-private LDS scratch
    lbf_t pw = (lbf_t)(ps[wave]);
#pragma unroll
    for (int nt2 = 0; nt2 < 4; ++nt2)
#pragma unroll
      for (int i = 0; i < 8; ++i)
        pw[(i + 8 * hi) * 64 + nt2 * 16 + n] = (bf16_t)st[nt2][i];

    v16bf pa0 = load_frag_l((lcbf_t)pw, 64, 0);
    v16bf pa1 = load_frag_l((lcbf_t)pw, 64, 32);
#pragma unroll
    for (int c2 = 0; c2 < 3; ++c2) {   // only real 48 output cols
      v16bf vb0 = load_frag_l((lcbf_t)vs + c2 * 16 * 64, 64, 0);
      v16bf vb1 = load_frag_l((lcbf_t)vs + c2 * 16 * 64, 64, 32);
      o[c2] = wmma_bf16(pa0, vb0, o[c2]);
      o[c2] = wmma_bf16(pa1, vb1, o[c2]);
    }
    __syncthreads();  // before next chunk overwrites ks/vs
  }

  // epilogue: O / l -> AO[b][t][h*48+col] bf16
#pragma unroll
  for (int c2 = 0; c2 < 3; ++c2) {
    const int col = c2 * 16 + n;
#pragma unroll
    for (int i = 0; i < 8; ++i) {
      const size_t t = (size_t)qrow + i + 8 * hi;
      ((gbf_t)AO)[((size_t)b * SEQ + t) * DIM + h * HDIM + col] =
          (bf16_t)(o[c2][i] / lrow[i]);
    }
  }
}

// ---------------- kernel 3: output projection + transposed store -------------
__global__ __launch_bounds__(256) void proj_kernel(
    const bf16_t* __restrict__ AO, const bf16_t* __restrict__ wob,
    const float* __restrict__ bo, float* __restrict__ out) {
  __shared__ __attribute__((aligned(16))) bf16_t as[64 * DIM];  // 48 KB
  const int b  = blockIdx.y;
  const int t0 = blockIdx.x * 64;
  const int tid = threadIdx.x;
  gcbf_t src = (gcbf_t)(AO + ((size_t)b * SEQ + t0) * DIM);
  lbf_t asl = (lbf_t)as;
  for (int e = tid * 8; e < 64 * DIM; e += 256 * 8)
    *(lv8bf_p)(asl + e) = *(gv8bf_cp)(src + e);
  __syncthreads();

  const int wave = tid >> 5, lane = tid & 31;
  const int n = lane & 15, hi = lane >> 4;
  for (int job = wave; job < 4 * 24; job += 8) {
    const int mt = job & 3;
    const int gn = (job >> 2) * 16;
    v8f acc = {0.f, 0.f, 0.f, 0.f, 0.f, 0.f, 0.f, 0.f};
#pragma unroll
    for (int kk = 0; kk < DIM; kk += 32) {
      v16bf a = load_frag_l((lcbf_t)as + mt * 16 * DIM, DIM, kk);
      v16bf w = load_frag_g((gcbf_t)(wob + (size_t)gn * DIM), DIM, kk);
      acc = wmma_bf16(a, w, acc);
    }
    const float bias = bo[gn + n];
    gf_t dst = (gf_t)(out + ((size_t)b * DIM + gn + n) * SEQ + t0 + mt * 16 +
                      8 * hi);
#pragma unroll
    for (int i = 0; i < 8; ++i) dst[i] = acc[i] + bias;  // coalesced along t
  }
}

// ---------------- launch ----------------
extern "C" void kernel_launch(void* const* d_in, const int* in_sizes, int n_in,
                              void* d_out, int out_size, void* d_ws, size_t ws_size,
                              hipStream_t stream) {
  const float* x  = (const float*)d_in[0];
  const float* wq = (const float*)d_in[1];
  const float* bq = (const float*)d_in[2];
  const float* wk = (const float*)d_in[3];
  const float* bk = (const float*)d_in[4];
  const float* wv = (const float*)d_in[5];
  const float* bv = (const float*)d_in[6];
  const float* wo = (const float*)d_in[7];
  const float* bo = (const float*)d_in[8];
  float* out = (float*)d_out;

  bf16_t* base = (bf16_t*)d_ws;
  const size_t QKVSZ = (size_t)BATCH * NH * SEQ * HDP;  // 4M bf16 each
  bf16_t* Qp  = base;
  bf16_t* Kp  = Qp + QKVSZ;
  bf16_t* VpT = Kp + QKVSZ;
  bf16_t* AO  = VpT + QKVSZ;
  bf16_t* wqb = AO + (size_t)BATCH * SEQ * DIM;
  bf16_t* wkb = wqb + (size_t)DIM * DIM;
  bf16_t* wvb = wkb + (size_t)DIM * DIM;
  bf16_t* wob = wvb + (size_t)DIM * DIM;

  // zero Q/K pads (zeroes whole Qp+Kp region; real cols overwritten by qkv)
  fill_zero_kernel<<<dim3(1024), dim3(256), 0, stream>>>(
      (unsigned int*)Qp, QKVSZ);  // 2*QKVSZ bf16 == QKVSZ dwords
  const int WN = DIM * DIM;
  cvt_kernel<<<dim3(144), dim3(256), 0, stream>>>(wq, wqb, WN);
  cvt_kernel<<<dim3(144), dim3(256), 0, stream>>>(wk, wkb, WN);
  cvt_kernel<<<dim3(144), dim3(256), 0, stream>>>(wv, wvb, WN);
  cvt_kernel<<<dim3(144), dim3(256), 0, stream>>>(wo, wob, WN);

  qkv_kernel<<<dim3(SEQ / 64, BATCH), dim3(256), 0, stream>>>(
      x, wqb, wkb, wvb, bq, bk, bv, Qp, Kp, VpT);
  attn_kernel<<<dim3(SEQ / 128, NH, BATCH), dim3(256), 0, stream>>>(
      Qp, Kp, VpT, AO);
  proj_kernel<<<dim3(SEQ / 64, BATCH), dim3(256), 0, stream>>>(
      AO, wob, bo, out);
}
